// DynamicConv2d_5471788335877
// MI455X (gfx1250) — matrix operand order repaired and checked by reading, not compiled, not run on hardware
//
#include <hip/hip_runtime.h>
#include <hip/hip_bf16.h>
#include <stdint.h>

typedef __bf16 bf16_t;
typedef __attribute__((ext_vector_type(16))) __bf16 v16bf;
typedef __attribute__((ext_vector_type(8)))  __bf16 v8bf;
typedef __attribute__((ext_vector_type(8)))  float  v8f;

#define BATCH 32
#define CIN   64
#define COUT  64
#define HH    128
#define WW    128
#define KTOT  576          // CIN*3*3
#define WROW  584          // LDS row stride for weights (mult of 8 -> 16B aligned b128)
#define XS_OFF 74752       // 64*584*2 bytes, 16B aligned
#define SMEM_BYTES (74752 + 6*130*64*2)   // 174,592 B < 320KB per WGP

// ---------------- Kernel 1: per-channel mean ----------------
__global__ __launch_bounds__(256)
void dc_mean_kernel(const float* __restrict__ x, float* __restrict__ feat) {
    __shared__ float red[256];
    const int bc = blockIdx.x;                    // b*CIN + ci
    const float* p = x + (size_t)bc * (HH * WW);
    float s = 0.f;
    for (int i = threadIdx.x; i < HH * WW; i += 256) s += p[i];
    red[threadIdx.x] = s;
    __syncthreads();
    for (int off = 128; off > 0; off >>= 1) {
        if (threadIdx.x < off) red[threadIdx.x] += red[threadIdx.x + off];
        __syncthreads();
    }
    if (threadIdx.x == 0) feat[bc] = red[0] * (1.0f / (HH * WW));
}

// ---------------- Kernel 2: h1 = relu(feat @ w1 + b1) ----------------
__global__ __launch_bounds__(256)
void dc_h1_kernel(const float* __restrict__ feat, const float* __restrict__ w1,
                  const float* __restrict__ b1, float* __restrict__ h1) {
    const int idx = blockIdx.x * 256 + threadIdx.x;   // 0..4095
    const int bb = idx >> 7, j = idx & 127;
    float s = b1[j];
    for (int ci = 0; ci < CIN; ++ci)
        s = fmaf(feat[bb * CIN + ci], w1[ci * 128 + j], s);
    h1[idx] = fmaxf(s, 0.f);
}

// ---------------- Kernel 3: wgen = h1 @ w2 + b2 (bf16, permuted layout) ------
// stored layout per sample: o = co*576 + (kh*3+kw)*64 + ci
__global__ __launch_bounds__(256)
void dc_wgen_kernel(const float* __restrict__ h1, const float* __restrict__ w2,
                    const float* __restrict__ b2, bf16_t* __restrict__ wg) {
    __shared__ float h1s[BATCH * 128];
    for (int i = threadIdx.x; i < BATCH * 128; i += 256) h1s[i] = h1[i];
    __syncthreads();
    const int cr = blockIdx.x * 256 + threadIdx.x;    // reference column 0..36863
    float s[BATCH];
    const float bias = b2[cr];
    #pragma unroll
    for (int bb = 0; bb < BATCH; ++bb) s[bb] = bias;
    for (int k = 0; k < 128; ++k) {
        const float wv = w2[(size_t)k * (COUT * KTOT) + cr];  // coalesced across lanes
        #pragma unroll
        for (int bb = 0; bb < BATCH; ++bb)
            s[bb] = fmaf(h1s[bb * 128 + k], wv, s[bb]);
    }
    // reference col -> (co, ci, r); destination o = co*576 + r*64 + ci
    const int co  = cr / KTOT;
    const int rem = cr - co * KTOT;
    const int ci  = rem / 9;
    const int r   = rem - ci * 9;
    const int o   = co * KTOT + r * 64 + ci;
    #pragma unroll
    for (int bb = 0; bb < BATCH; ++bb)
        wg[(size_t)bb * (COUT * KTOT) + o] = (bf16_t)s[bb];
}

// ---------------- Kernel 4: per-sample conv as implicit GEMM via WMMA --------
// grid = (32 row-tiles, 32 samples); block = 256 (8 wave32)
__global__ __launch_bounds__(256)
void dc_conv_wmma_kernel(const float* __restrict__ x, const bf16_t* __restrict__ wg,
                         float* __restrict__ out) {
    extern __shared__ char smem[];
    bf16_t* wsw = (bf16_t*)smem;                 // [64][584] bf16
    bf16_t* xs  = (bf16_t*)(smem + XS_OFF);      // [6 rows][130 cols][64 ci] bf16

    const int b   = blockIdx.y;
    const int h0  = blockIdx.x * 4;              // 4 output rows per WG
    const int tid = threadIdx.x;

    // ---- stage per-sample weights 64x576 -> LDS stride 584 (u32 copies) ----
    {
        const uint32_t* g = (const uint32_t*)(wg + (size_t)b * (COUT * KTOT));
        uint32_t* l = (uint32_t*)wsw;
        for (int i = tid; i < 64 * 288; i += 256) {
            const int co = i / 288, j = i - co * 288;
            l[co * 292 + j] = g[co * 288 + j];
        }
    }
    // ---- stage input rows h0-1..h0+4, cols -1..128, f32->bf16, halo=0 ------
    {
        const int NC = 130;
        for (int e = tid; e < 6 * 32 * NC; e += 256) {
            const int rr  = e / (32 * NC);
            const int rem = e - rr * 32 * NC;
            const int ci2 = rem / NC;            // pair of channels
            const int c   = rem - ci2 * NC;      // 0..129
            const int h   = h0 - 1 + rr;
            const int w   = c - 1;
            float f0 = 0.f, f1 = 0.f;
            if ((unsigned)h < HH && (unsigned)w < WW) {
                const size_t base = ((size_t)b * CIN + ci2 * 2) * (HH * WW)
                                    + (size_t)h * WW + w;
                f0 = x[base];
                f1 = x[base + HH * WW];
            }
            bf16_t b0 = (bf16_t)f0, b1v = (bf16_t)f1;
            uint16_t u0, u1;
            __builtin_memcpy(&u0, &b0, 2);
            __builtin_memcpy(&u1, &b1v, 2);
            ((uint32_t*)xs)[(rr * NC + c) * 32 + ci2] =
                ((uint32_t)u1 << 16) | (uint32_t)u0;
        }
    }
    __syncthreads();

    const int lane = tid & 31;
    const int wid  = tid >> 5;
    const int mi   = wid & 3;          // M-subtile (16 COUT rows)
    const int ngrp = wid >> 2;         // N-half (16 of 32 N-subtiles)
    const int sel  = lane >> 4;        // 0: K low half, 1: K high half
    const int ln   = lane & 15;

    // A fragment base: row co = mi*16+ln; lanes>=16 offset by 16 K-elements
    const bf16_t* arow = wsw + (mi * 16 + ln) * WROW + sel * 16;

    for (int p = 0; p < 4; ++p) {
        v8f acc0 = {}, acc1 = {}, acc2 = {}, acc3 = {};
        const int ns0 = ngrp * 16 + p * 4;
        const int hh0 = (ns0 + 0) >> 3, wb0 = ((ns0 + 0) & 7) * 16;
        const int hh1 = (ns0 + 1) >> 3, wb1 = ((ns0 + 1) & 7) * 16;
        const int hh2 = (ns0 + 2) >> 3, wb2 = ((ns0 + 2) & 7) * 16;
        const int hh3 = (ns0 + 3) >> 3, wb3 = ((ns0 + 3) & 7) * 16;

        for (int ks = 0; ks < 18; ++ks) {
            const int r   = ks >> 1;         // kh*3+kw
            const int cib = ks & 1;          // ci block of 32
            const int kh  = r / 3;
            const int kw  = r - kh * 3;
            const int kbase = r * 64 + cib * 32;

            // A fragment: K{0..15} (lanes 0-15) / K{16..31} (16-31)
            v16bf a;
            {
                union { v16bf v; v8bf h[2]; } u;
                u.h[0] = *(const v8bf*)(arow + kbase);
                u.h[1] = *(const v8bf*)(arow + kbase + 8);
                a = u.v;
            }
            const int cio = cib * 32 + sel * 16;   // lane's 16 contiguous ci

            {   // tile 0
                const bf16_t* bp = xs + (((hh0 + kh) * 130 + (wb0 + ln + kw)) * 64 + cio);
                union { v16bf v; v8bf h[2]; } u;
                u.h[0] = *(const v8bf*)bp; u.h[1] = *(const v8bf*)(bp + 8);
                acc0 = __builtin_amdgcn_wmma_f32_16x16x32_bf16(false, a, false, u.v,
                                                               (short)0, acc0, false, false);
            }
            {   // tile 1
                const bf16_t* bp = xs + (((hh1 + kh) * 130 + (wb1 + ln + kw)) * 64 + cio);
                union { v16bf v; v8bf h[2]; } u;
                u.h[0] = *(const v8bf*)bp; u.h[1] = *(const v8bf*)(bp + 8);
                acc1 = __builtin_amdgcn_wmma_f32_16x16x32_bf16(false, a, false, u.v,
                                                               (short)0, acc1, false, false);
            }
            {   // tile 2
                const bf16_t* bp = xs + (((hh2 + kh) * 130 + (wb2 + ln + kw)) * 64 + cio);
                union { v16bf v; v8bf h[2]; } u;
                u.h[0] = *(const v8bf*)bp; u.h[1] = *(const v8bf*)(bp + 8);
                acc2 = __builtin_amdgcn_wmma_f32_16x16x32_bf16(false, a, false, u.v,
                                                               (short)0, acc2, false, false);
            }
            {   // tile 3
                const bf16_t* bp = xs + (((hh3 + kh) * 130 + (wb3 + ln + kw)) * 64 + cio);
                union { v16bf v; v8bf h[2]; } u;
                u.h[0] = *(const v8bf*)bp; u.h[1] = *(const v8bf*)(bp + 8);
                acc3 = __builtin_amdgcn_wmma_f32_16x16x32_bf16(false, a, false, u.v,
                                                               (short)0, acc3, false, false);
            }
        }

        // ---- store: C/D layout: VGPR v -> M = v + sel*8, lane&15 -> N ----
        float* ob = out + (size_t)b * COUT * HH * WW;
        const int cobase = mi * 16;
        #pragma unroll
        for (int v = 0; v < 8; ++v) {
            const int m = v + sel * 8;
            ob[(size_t)(cobase + m) * (HH * WW) + (size_t)(h0 + hh0) * WW + wb0 + ln] = acc0[v];
            ob[(size_t)(cobase + m) * (HH * WW) + (size_t)(h0 + hh1) * WW + wb1 + ln] = acc1[v];
            ob[(size_t)(cobase + m) * (HH * WW) + (size_t)(h0 + hh2) * WW + wb2 + ln] = acc2[v];
            ob[(size_t)(cobase + m) * (HH * WW) + (size_t)(h0 + hh3) * WW + wb3 + ln] = acc3[v];
        }
    }
}

extern "C" void kernel_launch(void* const* d_in, const int* in_sizes, int n_in,
                              void* d_out, int out_size, void* d_ws, size_t ws_size,
                              hipStream_t stream) {
    const float* x  = (const float*)d_in[0];
    const float* w1 = (const float*)d_in[1];
    const float* b1 = (const float*)d_in[2];
    const float* w2 = (const float*)d_in[3];
    const float* b2 = (const float*)d_in[4];
    float* out = (float*)d_out;

    char* ws = (char*)d_ws;
    float*  feat = (float*)ws;                       //  2048 f32
    float*  h1   = (float*)(ws + 8192);              //  4096 f32
    bf16_t* wg   = (bf16_t*)(ws + 8192 + 16384);     // 32*36864 bf16 (~2.3 MB)

    dc_mean_kernel<<<BATCH * CIN, 256, 0, stream>>>(x, feat);
    dc_h1_kernel<<<16, 256, 0, stream>>>(feat, w1, b1, h1);
    dc_wgen_kernel<<<(COUT * KTOT) / 256, 256, 0, stream>>>(h1, w2, b2, wg);

    dim3 grid(HH / 4, BATCH);
    dc_conv_wmma_kernel<<<grid, 256, SMEM_BYTES, stream>>>(x, wg, out);
}
